// LigerFusedLinearJSDLoss_10737418240013
// MI455X (gfx1250) — compile-verified, matrix operand order checked by reading, not run on hardware
//
#include <hip/hip_runtime.h>
#include <hip/hip_bf16.h>

// ---------------------------------------------------------------------------
// Fused Linear JSD loss for MI455X (gfx1250, wave32, WMMA bf16 path).
//
// Tiling: 256-thread WG (8 wave32s as 4x2) computes a 128x64 logit tile.
// A (activations) and B (weights) K-chunks are staged to LDS as bf16 with
// DOUBLE BUFFERING: chunk i+1's global loads are issued as one clause
// (static per-thread quad assignment, fully unrolled) and converted/stored
// while chunk i feeds ds_load_b128 + v_wmma_f32_16x16x32_bf16; one barrier
// per chunk. blockIdx.x = row-block (fastest) so 16 WGs reuse each 64-col
// weight slab straight out of L2; weights stream from HBM ~once per pass.
//
// Pass 1 (x2): online-softmax partials (m, s) per (row, 64-col tile).
// Pass 2 (x2): reduce tile partials -> per-row logsumexp.
// Pass 3:      recompute student+teacher logit tiles together, accumulate
//              JSD + label-CE, atomicAdd the scalar loss.
// ---------------------------------------------------------------------------

#define N_TOK   2048
#define VOCAB   32000
#define H_STU   2048
#define H_TEA   4096
#define MR      128                 // rows per WG
#define NR      64                  // cols per WG
#define NT      (VOCAB / NR)        // 500 col tiles
#define KC      64                  // K chunk staged in LDS
#define SA      (KC + 16)           // LDS row stride (ushort) -> 160B, 16B aligned
#define SB      (KC + 16)
#define LDS_STRIDE (MR * SA + NR * SB)   // ushorts per buffer (30720 B)
#define WG_THR  256                 // 8 wave32s
#define IGN_IDX (-100)

typedef __attribute__((ext_vector_type(16))) __bf16 v16bf;
typedef __attribute__((ext_vector_type(8)))  float  v8f;

union Frag { v16bf v; uint4 q[2]; };

// f32 -> bf16 bits, round-to-nearest-even
__device__ __forceinline__ unsigned int bf1(float f) {
    unsigned int u = __float_as_uint(f);
    return (u + 0x7FFFu + ((u >> 16) & 1u)) >> 16;
}
__device__ __forceinline__ unsigned int pk2(float a, float b) {
    return bf1(a) | (bf1(b) << 16);
}

__device__ __forceinline__ void ons_merge(float& m, float& s, float om, float os) {
    float mn = fmaxf(m, om);
    s = s * __expf(m - mn) + os * __expf(om - mn);
    m = mn;
}

__device__ __forceinline__ v8f wmma_bf16(const Frag& a, const Frag& b, v8f c) {
    return __builtin_amdgcn_wmma_f32_16x16x32_bf16(false, a.v, false, b.v,
                                                   (short)0, c, false, false);
}

__device__ __forceinline__ Frag lds_frag(const unsigned short* p) {
    Frag f;
    f.q[0] = *(const uint4*)p;
    f.q[1] = *(const uint4*)(p + 16);
    return f;
}

// Stage a K-chunk of A (MR rows) and B (NR cols) into one LDS buffer as bf16.
// Static assignment: thread owns quad (tid&15) of rows/cols (tid>>4) + 16k.
// All global loads are issued first (one clause), then convert + ds_store.
template <int H>
__device__ __forceinline__ void stage_chunk(const float* __restrict__ X, int rowbase,
                                            const float* __restrict__ W, int colbase,
                                            int kc0, unsigned short* ldsA,
                                            unsigned short* ldsB, int tid) {
    const int q  = (tid & 15) * 4;      // float offset inside the K chunk
    const int r0 = tid >> 4;            // 0..15

    float4 fa[8];
#pragma unroll
    for (int k = 0; k < 8; ++k)
        fa[k] = *(const float4*)(X + (size_t)(rowbase + r0 + 16 * k) * H + kc0 + q);

    float4 fb[4];
#pragma unroll
    for (int k = 0; k < 4; ++k)
        fb[k] = *(const float4*)(W + (size_t)(colbase + r0 + 16 * k) * H + kc0 + q);

    if (kc0 + KC < H) {                 // pull next weight chunk toward L2
#pragma unroll
        for (int k = 0; k < 4; ++k)
            __builtin_prefetch(W + (size_t)(colbase + r0 + 16 * k) * H + kc0 + KC + q, 0, 1);
    }

#pragma unroll
    for (int k = 0; k < 8; ++k)
        *(uint2*)(ldsA + (r0 + 16 * k) * SA + q) =
            make_uint2(pk2(fa[k].x, fa[k].y), pk2(fa[k].z, fa[k].w));
#pragma unroll
    for (int k = 0; k < 4; ++k)
        *(uint2*)(ldsB + (r0 + 16 * k) * SB + q) =
            make_uint2(pk2(fb[k].x, fb[k].y), pk2(fb[k].z, fb[k].w));
}

// Double-buffered GEMM over K accumulating into acc[2][2] (wave tile 32x32).
template <int H>
__device__ __forceinline__ void gemm_tile(const float* __restrict__ X, int rowbase,
                                          const float* __restrict__ W, int colbase,
                                          unsigned short* smem, int tid, int wr, int wc,
                                          int lrow, int koff, v8f acc[2][2]) {
    stage_chunk<H>(X, rowbase, W, colbase, 0, smem, smem + MR * SA, tid);
    __syncthreads();
#pragma unroll 1
    for (int kc0 = 0; kc0 < H; kc0 += KC) {
        const int cur = (kc0 / KC) & 1;
        unsigned short* ldsA = smem + cur * LDS_STRIDE;
        unsigned short* ldsB = ldsA + MR * SA;
        // stage next chunk into the idle buffer while this one computes
        if (kc0 + KC < H) {
            unsigned short* nA = smem + (cur ^ 1) * LDS_STRIDE;
            stage_chunk<H>(X, rowbase, W, colbase, kc0 + KC, nA, nA + MR * SA, tid);
        }
#pragma unroll
        for (int kk = 0; kk < KC; kk += 32) {
            Frag a0 = lds_frag(ldsA + (wr * 32 + 0  + lrow) * SA + kk + koff);
            Frag a1 = lds_frag(ldsA + (wr * 32 + 16 + lrow) * SA + kk + koff);
            Frag b0 = lds_frag(ldsB + (wc * 32 + 0  + lrow) * SB + kk + koff);
            Frag b1 = lds_frag(ldsB + (wc * 32 + 16 + lrow) * SB + kk + koff);
            acc[0][0] = wmma_bf16(a0, b0, acc[0][0]);
            acc[0][1] = wmma_bf16(a0, b1, acc[0][1]);
            acc[1][0] = wmma_bf16(a1, b0, acc[1][0]);
            acc[1][1] = wmma_bf16(a1, b1, acc[1][1]);
        }
        __syncthreads();   // next buffer fully staged & this buffer fully read
    }
}

// ------------------------------ Pass 1 -------------------------------------
template <int H>
__global__ void fljsd_stats(const float* __restrict__ X, const float* __restrict__ W,
                            float2* __restrict__ partial) {
    extern __shared__ unsigned short smem[];
    float2* red = (float2*)(smem + 2 * LDS_STRIDE);   // 128 rows x 2 col-halves

    const int tid = threadIdx.x;
    const int rowbase = blockIdx.x * MR;
    const int colbase = blockIdx.y * NR;
    const int wave = tid >> 5, lane = tid & 31;
    const int lrow = lane & 15, half = lane >> 4, koff = half * 8;
    const int wr = wave >> 1, wc = wave & 1;

    v8f acc[2][2] = {};
    gemm_tile<H>(X, rowbase, W, colbase, smem, tid, wr, wc, lrow, koff, acc);

    // online-softmax partial over this wave's 32x32 tile
#pragma unroll
    for (int mt = 0; mt < 2; ++mt) {
#pragma unroll
        for (int j = 0; j < 8; ++j) {
            float x0 = acc[mt][0][j], x1 = acc[mt][1][j];
            float m = fmaxf(x0, x1);
            float s = __expf(x0 - m) + __expf(x1 - m);
#pragma unroll
            for (int off = 8; off; off >>= 1) {
                float om = __shfl_xor(m, off);
                float os = __shfl_xor(s, off);
                ons_merge(m, s, om, os);
            }
            if (lrow == 0)
                red[(wr * 32 + mt * 16 + half * 8 + j) * 2 + wc] = make_float2(m, s);
        }
    }
    __syncthreads();
    if (tid < MR) {
        float2 p0 = red[tid * 2], p1 = red[tid * 2 + 1];
        float m = p0.x, s = p0.y;
        ons_merge(m, s, p1.x, p1.y);
        partial[(size_t)(rowbase + tid) * NT + blockIdx.y] = make_float2(m, s);
    }
}

// ------------------------------ Pass 2 -------------------------------------
__global__ void fljsd_lse(const float2* __restrict__ partial, float* __restrict__ lse) {
    int r = blockIdx.x * blockDim.x + threadIdx.x;
    if (r >= N_TOK) return;
    float m = -__builtin_inff(), s = 0.f;
    for (int c = 0; c < NT; ++c) {
        float2 p = partial[(size_t)r * NT + c];
        ons_merge(m, s, p.x, p.y);
    }
    lse[r] = m + __logf(s);
}

// ------------------------------ Pass 3 -------------------------------------
__global__ void fljsd_loss(const float* __restrict__ Xs, const float* __restrict__ Ws,
                           const float* __restrict__ Xt, const float* __restrict__ Wt,
                           const float* __restrict__ lse_s, const float* __restrict__ lse_t,
                           const int* __restrict__ labels, float* __restrict__ out) {
    extern __shared__ unsigned short smem[];

    const int tid = threadIdx.x;
    const int rowbase = blockIdx.x * MR;
    const int colbase = blockIdx.y * NR;
    const int wave = tid >> 5, lane = tid & 31;
    const int lrow = lane & 15, half = lane >> 4, koff = half * 8;
    const int wr = wave >> 1, wc = wave & 1;

    v8f cs[2][2] = {};
    gemm_tile<H_STU>(Xs, rowbase, Ws, colbase, smem, tid, wr, wc, lrow, koff, cs);
    v8f ct[2][2] = {};
    gemm_tile<H_TEA>(Xt, rowbase, Wt, colbase, smem, tid, wr, wc, lrow, koff, ct);

    const float LOG_HALF = -0.6931471805599453f;
    float acc = 0.f;
#pragma unroll
    for (int mt = 0; mt < 2; ++mt) {
#pragma unroll
        for (int j = 0; j < 8; ++j) {
            const int r = rowbase + wr * 32 + mt * 16 + half * 8 + j;
            const float ls = lse_s[r];
            const float lt = lse_t[r];
            const int   lb = labels[r];
            const float mv = (lb != IGN_IDX) ? 1.0f : 0.0f;
#pragma unroll
            for (int nt = 0; nt < 2; ++nt) {
                const int col = colbase + wc * 32 + nt * 16 + lrow;
                float slp = cs[mt][nt][j] - ls;       // student log-prob
                float tlp = ct[mt][nt][j] - lt;       // teacher log-prob
                float mx  = fmaxf(slp, tlp);
                float lm  = LOG_HALF + mx + __logf(__expf(slp - mx) + __expf(tlp - mx));
                float jsd = 0.5f * __expf(tlp) * (tlp - lm)
                          + 0.5f * __expf(slp) * (slp - lm);
                acc += mv * jsd;
                if (col == lb) acc -= slp;            // CE = lse_s - s_logit
            }
        }
    }
#pragma unroll
    for (int off = 16; off; off >>= 1) acc += __shfl_xor(acc, off);
    if (lane == 0) atomicAdd(out, acc * (0.5f / (float)N_TOK));
}

// ---------------------------------------------------------------------------
extern "C" void kernel_launch(void* const* d_in, const int* in_sizes, int n_in,
                              void* d_out, int out_size, void* d_ws, size_t ws_size,
                              hipStream_t stream) {
    const float* Xs     = (const float*)d_in[0];   // (2048, 2048)
    const float* Ws     = (const float*)d_in[1];   // (32000, 2048)
    const float* Xt     = (const float*)d_in[2];   // (2048, 4096)
    const float* Wt     = (const float*)d_in[3];   // (32000, 4096)
    const int*   labels = (const int*)d_in[4];     // (2048,)
    float* out          = (float*)d_out;

    float2* part_s = (float2*)d_ws;                         // 2048*500 float2 = 8 MB
    float2* part_t = part_s + (size_t)N_TOK * NT;           // 8 MB
    float*  lse_s  = (float*)(part_t + (size_t)N_TOK * NT);
    float*  lse_t  = lse_s + N_TOK;

    hipMemsetAsync(out, 0, sizeof(float), stream);

    dim3 grid(N_TOK / MR, NT);        // x = row-block (fastest) -> weight slab reuse
    dim3 block(WG_THR);

    const size_t ldsGemm = (size_t)2 * LDS_STRIDE * sizeof(unsigned short); // 61440 B
    const size_t lds1 = ldsGemm + (size_t)MR * sizeof(float2);

    fljsd_stats<H_STU><<<grid, block, lds1, stream>>>(Xs, Ws, part_s);
    fljsd_stats<H_TEA><<<grid, block, lds1, stream>>>(Xt, Wt, part_t);

    fljsd_lse<<<dim3((N_TOK + 255) / 256), dim3(256), 0, stream>>>(part_s, lse_s);
    fljsd_lse<<<dim3((N_TOK + 255) / 256), dim3(256), 0, stream>>>(part_t, lse_t);

    fljsd_loss<<<grid, block, ldsGemm, stream>>>(Xs, Ws, Xt, Wt, lse_s, lse_t, labels, out);
}
